// SparseQuantLinear_64330020159883
// MI455X (gfx1250) — matrix operand order repaired and checked by reading, not compiled, not run on hardware
//
#include <hip/hip_runtime.h>

// ---------------------------------------------------------------------------
// SINQ sparse-quant linear for MI455X (gfx1250, wave32).
//   Pass 0: x fp32 -> bf16 xb[M,K] in ws           (done once, amortized 32x)
//   Pass 1: dequant W_q -> bf16 Wb[O,K] in ws      (mask + scale2 folded in)
//   Pass 2: bf16 WMMA GEMM out = x * W^T + bias, double-buffered LDS fed by
//           GLOBAL_LOAD_ASYNC_TO_LDS_B128 (ASYNCcnt-tracked DMA).
// GEMM: 128x128x64 block tile, 4 wave32s in 2x2 grid, 64x64 per wave
//       = 4x4 WMMA tiles -> 16 v_wmma per k-step vs 16 ds_load_b128 (1:1).
// Workspace: Wb = O*K*2 (32 MB) + xb = M*K*2 (134 MB).
// ---------------------------------------------------------------------------

typedef __attribute__((ext_vector_type(16))) __bf16        bf16x16;
typedef __attribute__((ext_vector_type(8)))  float         floatx8;
typedef __attribute__((ext_vector_type(4)))  float         floatx4;
typedef __attribute__((ext_vector_type(4)))  int           intx4;
typedef __attribute__((ext_vector_type(4)))  unsigned int  uintx4;
typedef int v4i_vs __attribute__((vector_size(16)));       // builtin's pointee type

#ifdef __has_builtin
#define CD5_HAS(X) __has_builtin(X)
#else
#define CD5_HAS(X) 0
#endif

#if CD5_HAS(__builtin_amdgcn_global_load_async_to_lds_b128)
#define CD5_ASYNC_LDS 1
#else
#define CD5_ASYNC_LDS 0
#endif

__device__ __forceinline__ unsigned int f2bf_bits(float f) {
    unsigned int u = __builtin_bit_cast(unsigned int, f);
    return (u + 0x7FFFu + ((u >> 16) & 1u)) >> 16;   // round-to-nearest-even
}
__device__ __forceinline__ unsigned int pack2bf(float lo, float hi) {
    return f2bf_bits(lo) | (f2bf_bits(hi) << 16);
}

// 16-byte global -> LDS copy; async DMA path on gfx1250 when builtin exists.
__device__ __forceinline__ void cp16_g2l(unsigned short* dst_lds,
                                         const unsigned short* src_glb) {
#if CD5_ASYNC_LDS
    __builtin_amdgcn_global_load_async_to_lds_b128(
        (__attribute__((address_space(1))) v4i_vs*)(src_glb),
        (__attribute__((address_space(3))) v4i_vs*)(dst_lds),
        0, 0);
#else
    *(uintx4*)dst_lds = *(const uintx4*)src_glb;
#endif
}

#if CD5_ASYNC_LDS
#if CD5_HAS(__builtin_amdgcn_s_wait_asynccnt)
#define CD5_WAIT_ASYNC(n) __builtin_amdgcn_s_wait_asynccnt(n)
#else
#define CD5_WAIT_ASYNC(n) asm volatile("s_wait_asynccnt %0" :: "i"(n) : "memory")
#endif
#else
#define CD5_WAIT_ASYNC(n) ((void)0)   // sync path: barrier handles dscnt
#endif

// ---------------------------------------------------------------------------
// Pass 0: x fp32 -> bf16 (RNE), 8 elements/thread.
// ---------------------------------------------------------------------------
__global__ __launch_bounds__(256)
void x_to_bf16(const float* __restrict__ x, unsigned short* __restrict__ xb) {
    size_t base = ((size_t)blockIdx.x * 256 + threadIdx.x) * 8;
    floatx4 a = *(const floatx4*)(x + base);
    floatx4 b = *(const floatx4*)(x + base + 4);
    uintx4 v;
    v.x = pack2bf(a.x, a.y); v.y = pack2bf(a.z, a.w);
    v.z = pack2bf(b.x, b.y); v.w = pack2bf(b.z, b.w);
    *(uintx4*)(xb + base) = v;
}

// ---------------------------------------------------------------------------
// Pass 1: W = ((W_q - zero) * scale) * scale2 masked -> bf16, row-major [O,K].
// ---------------------------------------------------------------------------
__global__ __launch_bounds__(256)
void sinq_dequant_bf16(const int* __restrict__ Wq,
                       const float* __restrict__ scales,
                       const float* __restrict__ zeros,
                       const unsigned char* __restrict__ mask,
                       const float* __restrict__ scale2,
                       unsigned short* __restrict__ Wb,
                       int I, int Ig, int G) {
    size_t t    = (size_t)blockIdx.x * blockDim.x + threadIdx.x;
    size_t base = t * 8;
    int o   = (int)(base / (size_t)I);
    int i0  = (int)(base % (size_t)I);
    int grp = i0 / G;

    float sc = scales[(size_t)o * Ig + grp];
    float zp = zeros [(size_t)o * Ig + grp];

    intx4 q0 = ((const intx4*)(Wq + base))[0];
    intx4 q1 = ((const intx4*)(Wq + base))[1];
    floatx4 s2a = *(const floatx4*)(scale2 + i0);
    floatx4 s2b = *(const floatx4*)(scale2 + i0 + 4);
    unsigned long long mb = *(const unsigned long long*)(mask + base);

    int   q[8]  = {q0.x, q0.y, q0.z, q0.w, q1.x, q1.y, q1.z, q1.w};
    float s2[8] = {s2a.x, s2a.y, s2a.z, s2a.w, s2b.x, s2b.y, s2b.z, s2b.w};

    unsigned int pk[4];
#pragma unroll
    for (int j = 0; j < 4; ++j) {
        float lo = ((float)q[2 * j]     - zp) * sc * s2[2 * j];
        float hi = ((float)q[2 * j + 1] - zp) * sc * s2[2 * j + 1];
        if (!((mb >> (16 * j))     & 0xFFull)) lo = 0.0f;
        if (!((mb >> (16 * j + 8)) & 0xFFull)) hi = 0.0f;
        pk[j] = pack2bf(lo, hi);
    }
    uintx4 v; v.x = pk[0]; v.y = pk[1]; v.z = pk[2]; v.w = pk[3];
    *(uintx4*)(Wb + base) = v;
}

// ---------------------------------------------------------------------------
// Pass 2: bf16 WMMA GEMM, 4 waves, 64x64 per wave, async double-buffered LDS.
// ---------------------------------------------------------------------------
#define BM  128
#define BN  128
#define BK  64
#define PAD 8     // halves; 144B row stride: 16B-aligned rows, bank-skewed

__global__ __launch_bounds__(128)
void sinq_wmma_gemm(const unsigned short* __restrict__ xb,   // [M,K] bf16
                    const unsigned short* __restrict__ Wb,   // [N,K] bf16
                    const float* __restrict__ bias,          // [N]
                    float* __restrict__ out,                 // [M,N] fp32
                    int M, int N, int K) {
    __shared__ unsigned short As[2][BM][BK + PAD];
    __shared__ unsigned short Bs[2][BN][BK + PAD];

    const int tid  = threadIdx.x;         // 0..127 (4 wave32s)
    const int lane = tid & 31;
    const int wid  = tid >> 5;            // 0..3
    const int wmb  = (wid >> 1) * 64;     // wave M base in tile (2x2 wave grid)
    const int wnb  = (wid & 1) * 64;      // wave N base in tile

    const int bm = blockIdx.y * BM;
    const int bn = blockIdx.x * BN;

    // Staging map: 128 threads x (8 A + 8 B) 16-byte async copies per stage.
    const int c8 = tid & 7;               // 16-byte (8-half) column within BK
    const int r0 = tid >> 3;              // 0..15, step 16 over 128 rows

    auto stage = [&](int buf, int kt) {
        const int k0 = kt * BK;
#pragma unroll
        for (int j = 0; j < 8; ++j) {
            const int r = r0 + j * 16;
            cp16_g2l(&As[buf][r][c8 * 8],
                     xb + (size_t)(bm + r) * K + k0 + c8 * 8);
            cp16_g2l(&Bs[buf][r][c8 * 8],
                     Wb + (size_t)(bn + r) * K + k0 + c8 * 8);
        }
    };

    floatx8 acc[4][4];
#pragma unroll
    for (int i = 0; i < 4; ++i)
#pragma unroll
        for (int j = 0; j < 4; ++j)
#pragma unroll
            for (int e = 0; e < 8; ++e) acc[i][j][e] = 0.0f;

    const int nk = K / BK;
    stage(0, 0);                               // prime the pipe (16 async ops)

    for (int kt = 0; kt < nk; ++kt) {
        const int cur = kt & 1;
        if (kt + 1 < nk) {
            stage(cur ^ 1, kt + 1);            // issue next stage (16 more)
            CD5_WAIT_ASYNC(16);                // in-order: stage kt has landed
        } else {
            CD5_WAIT_ASYNC(0);
        }
        __syncthreads();                       // everyone's stage-kt in LDS

#pragma unroll
        for (int ks = 0; ks < 2; ++ks) {
            // CDNA5 16-bit A layout: lanes 0-15 hold K {0..7,16..23},
            // lanes 16-31 hold K {8..15,24..31} of the 16x32 fragment.
            const int ak = ks * 32 + ((lane >> 4) << 3);   // 0|8
            // B 32x16: lane = n; lanes 0-15 hold K {0..7,16..23}, 16-31 K {8..15,24..31}.
            const int bk = ks * 32 + ((lane >> 4) << 3);   // 0|8

            union Frag { bf16x16 v; uintx4 u[2]; } a[4], b[4];
#pragma unroll
            for (int mt = 0; mt < 4; ++mt) {
                const unsigned short* p = &As[cur][wmb + mt * 16 + (lane & 15)][0];
                a[mt].u[0] = *(const uintx4*)(p + ak);
                a[mt].u[1] = *(const uintx4*)(p + ak + 16);
            }
#pragma unroll
            for (int nt = 0; nt < 4; ++nt) {
                const unsigned short* p = &Bs[cur][wnb + nt * 16 + (lane & 15)][0];
                b[nt].u[0] = *(const uintx4*)(p + bk);
                b[nt].u[1] = *(const uintx4*)(p + bk + 16);
            }
#pragma unroll
            for (int mt = 0; mt < 4; ++mt)
#pragma unroll
                for (int nt = 0; nt < 4; ++nt)
                    acc[mt][nt] = __builtin_amdgcn_wmma_f32_16x16x32_bf16(
                        false, a[mt].v, false, b[nt].v,
                        (short)0, acc[mt][nt], false, false);
        }
        __syncthreads();   // compute done before next stage overwrites buffer
    }

    // Epilogue: C/D layout — VGPR r: lanes 0-15 -> M=r, lanes 16-31 -> M=8+r.
#pragma unroll
    for (int nt = 0; nt < 4; ++nt) {
        const int   n  = bn + wnb + nt * 16 + (lane & 15);
        const float bv = bias[n];
#pragma unroll
        for (int mt = 0; mt < 4; ++mt) {
            const int m0 = bm + wmb + mt * 16 + ((lane >> 4) << 3);
#pragma unroll
            for (int r = 0; r < 8; ++r)
                out[(size_t)(m0 + r) * N + n] = acc[mt][nt][r] + bv;
        }
    }
}

// ---------------------------------------------------------------------------
// Host launcher. Inputs: x, W_q, scales, zeros, mask, scale2, bias, group_size
// ---------------------------------------------------------------------------
extern "C" void kernel_launch(void* const* d_in, const int* in_sizes, int n_in,
                              void* d_out, int out_size, void* d_ws, size_t ws_size,
                              hipStream_t stream) {
    const float*         x      = (const float*)d_in[0];
    const int*           Wq     = (const int*)d_in[1];
    const float*         scales = (const float*)d_in[2];
    const float*         zeros  = (const float*)d_in[3];
    const unsigned char* mask   = (const unsigned char*)d_in[4];  // jnp.bool_ = 1B
    const float*         scale2 = (const float*)d_in[5];
    const float*         bias   = (const float*)d_in[6];
    float*               out    = (float*)d_out;

    const long long I = in_sizes[5];                       // K = 4096
    const long long O = in_sizes[6];                       // N = 4096
    const long long G = (O * I) / (long long)in_sizes[2];  // group size = 128
    const long long M = (long long)in_sizes[0] / I;        // B*S = 16384

    unsigned short* Wb = (unsigned short*)d_ws;            // O*K bf16
    unsigned short* xb = Wb + (size_t)O * I;               // M*K bf16

    // Pass 0: x -> bf16
    {
        long long threads = (M * I) / 8;
        dim3 grid((unsigned)((threads + 255) / 256));
        x_to_bf16<<<grid, 256, 0, stream>>>(x, xb);
    }
    // Pass 1: dequant -> bf16 weights
    {
        long long threads = (O * I) / 8;
        dim3 grid((unsigned)((threads + 255) / 256));
        sinq_dequant_bf16<<<grid, 256, 0, stream>>>(
            Wq, scales, zeros, mask, scale2, Wb, (int)I, (int)(I / G), (int)G);
    }
    // Pass 2: WMMA GEMM + bias (128 threads = 4 wave32s per block)
    {
        dim3 grid((unsigned)(O / BN), (unsigned)(M / BM));
        sinq_wmma_gemm<<<grid, 128, 0, stream>>>(
            xb, Wb, bias, out, (int)M, (int)O, (int)I);
    }
}
